// Attention_10969346474050
// MI455X (gfx1250) — compile-verified, matrix operand order checked by reading, not run on hardware
//
#include <hip/hip_runtime.h>
#include <hip/hip_bf16.h>

typedef __attribute__((ext_vector_type(16))) _Float16 v16h;
typedef __attribute__((ext_vector_type(8)))  _Float16 v8h;
typedef __attribute__((ext_vector_type(4)))  _Float16 v4h;
typedef __attribute__((ext_vector_type(8)))  float    v8f;
typedef __attribute__((ext_vector_type(4)))  float    v4f;
typedef __attribute__((ext_vector_type(4)))  unsigned int v4u;
typedef __attribute__((ext_vector_type(8)))  int      v8i;
typedef __attribute__((ext_vector_type(4)))  int      v4i;

#define Bn   16
#define Nn   4096
#define Cn   1024
#define Hn   16
#define Dn   64
#define SCALE 0.125f   // 64^-0.5

#if defined(__has_builtin)
#if __has_builtin(__builtin_amdgcn_tensor_load_to_lds) && __has_builtin(__builtin_amdgcn_s_wait_tensorcnt)
#define USE_TDM 1
#endif
#endif
#ifndef USE_TDM
#define USE_TDM 0
#endif

static __device__ inline v8f wmma_f16(v16h a, v16h b, v8f c) {
  return __builtin_amdgcn_wmma_f32_16x16x32_f16(false, a, false, b, (short)0, c,
                                                false, false);
}

// build a 16-half fragment from two contiguous 8-half chunks (p and p+16)
static __device__ inline v16h mk16(const _Float16* p) {
  v8h lo = *(const v8h*)p;
  v8h hi = *(const v8h*)(p + 16);
  v16h r;
#pragma unroll
  for (int i = 0; i < 8; ++i) { r[i] = lo[i]; r[i + 8] = hi[i]; }
  return r;
}

static __device__ inline v4h cvt4(v4f f) {
  v4h h;
  h[0] = (_Float16)f[0]; h[1] = (_Float16)f[1];
  h[2] = (_Float16)f[2]; h[3] = (_Float16)f[3];
  return h;
}

#if USE_TDM
// Issue one TDM 2D tile load: tile_rows rows of tile_k halfs (f16), global row
// stride row_stride_halfs, packed contiguously into LDS at lds_byte_addr.
// D# layout per CDNA5 ISA ch.8: group0 = {count/flags, lds_addr, gaddr_lo,
// gaddr_hi|type=2}, group1 = {data_size=2B, tensor dims (huge, no clip),
// tile dims, dim0 stride}, remaining groups unused (2D tile) -> zeros.
// This toolchain exposes the 6-arg builtin: (v4u, v8i, v4i, v4i, v8i, cpol).
static __device__ inline void tdm_load_tile(unsigned lds_byte_addr,
                                            unsigned long long gaddr,
                                            unsigned tile_k_halfs,
                                            unsigned tile_rows,
                                            unsigned row_stride_halfs) {
  v4u g0;
  g0[0] = 1u;                                   // count=1, user descriptor
  g0[1] = lds_byte_addr;                        // lds_addr
  g0[2] = (unsigned)gaddr;                      // global_addr[31:0]
  g0[3] = (unsigned)(gaddr >> 32) | (2u << 30); // global_addr[56:32] | type=2
  v8i g1;
  g1[0] = (int)(1u << 16);                      // data_size=1 -> 2 bytes
  g1[1] = 0;                                    // tensor_dim0 = 1<<30 (lo16=0)
  g1[2] = (int)0x4000u;                         // dim0 hi16 | dim1 lo16 (dim1=1<<30)
  g1[3] = (int)(0x4000u | (tile_k_halfs << 16));// dim1 hi16 | tile_dim0
  g1[4] = (int)tile_rows;                       // tile_dim1 | tile_dim2=0
  g1[5] = (int)row_stride_halfs;                // tensor_dim0_stride lo32
  g1[6] = 0;
  g1[7] = 0;
  v4i gz4 = {0, 0, 0, 0};
  v8i gz8 = {0, 0, 0, 0, 0, 0, 0, 0};
  __builtin_amdgcn_tensor_load_to_lds(g0, g1, gz4, gz4, gz8, 0);
}
#endif

// ---------------------------------------------------------------- q = fore@Wq+bq
__global__ __launch_bounds__(256) void qproj_kernel(
    const float* __restrict__ fore, const float* __restrict__ Wq,
    const float* __restrict__ bq, float* __restrict__ q) {
  int gid = blockIdx.x * 256 + threadIdx.x;   // 16*1024
  int b = gid >> 10;
  int c = gid & 1023;
  float sum = bq[c];
  const float* fr = fore + b * Cn;
#pragma unroll 4
  for (int i = 0; i < Cn; ++i) sum = fmaf(fr[i], Wq[(size_t)i * Cn + c], sum);
  q[gid] = sum;
}

// -------------------------------------------- u_t[(b*16+h)*1024+c] = scale*Wk[c,h,:].q
__global__ __launch_bounds__(256) void uproj_kernel(
    const float* __restrict__ Wk, const float* __restrict__ q,
    _Float16* __restrict__ u_t) {
  int gid = blockIdx.x * 256 + threadIdx.x;   // 16*16*1024
  int b  = gid >> 14;
  int h  = (gid >> 10) & 15;
  int ck = gid & 1023;
  const float* wr = Wk + (size_t)ck * Cn + h * Dn;
  const float* qr = q + b * Cn + h * Dn;
  float sum = 0.f;
#pragma unroll 8
  for (int d = 0; d < Dn; ++d) sum = fmaf(wr[d], qr[d], sum);
  u_t[gid] = (_Float16)(SCALE * sum);
}

__global__ __launch_bounds__(256) void betaproj_kernel(
    const float* __restrict__ bk, const float* __restrict__ q,
    float* __restrict__ beta) {
  int idx = threadIdx.x;          // 256 = B*H
  int b = idx >> 4, h = idx & 15;
  const float* br = bk + h * Dn;
  const float* qr = q + b * Cn + h * Dn;
  float sum = 0.f;
#pragma unroll 8
  for (int d = 0; d < Dn; ++d) sum = fmaf(br[d], qr[d], sum);
  beta[idx] = SCALE * sum;
}

// -------------------------------------------- Wvt[c][k] = (f16) Wv[k][c]
__global__ __launch_bounds__(256) void wvt_kernel(
    const float* __restrict__ Wv, _Float16* __restrict__ Wvt) {
  __shared__ float t[32][33];
  int tx = threadIdx.x & 31;
  int ty = threadIdx.x >> 5;          // 0..7
  int c0 = blockIdx.x * 32;           // Wv column block
  int k0 = blockIdx.y * 32;           // Wv row block
#pragma unroll
  for (int i = 0; i < 4; ++i)
    t[ty + i * 8][tx] = Wv[(size_t)(k0 + ty + i * 8) * Cn + c0 + tx];
  __syncthreads();
#pragma unroll
  for (int i = 0; i < 4; ++i)
    Wvt[(size_t)(c0 + ty + i * 8) * Cn + k0 + tx] = (_Float16)t[tx][ty + i * 8];
}

// -------------------------------------------- scores s[(b*16+h)*4096+n] = x.u + beta
#define USTRIDE 1032
__global__ __launch_bounds__(256) void score_kernel(
    const float* __restrict__ x, const _Float16* __restrict__ u_t,
    const float* __restrict__ beta, float* __restrict__ s) {
  __shared__ __attribute__((aligned(16))) _Float16 Us[16 * USTRIDE];

  int tid = threadIdx.x;
  int rowBase = blockIdx.x * 128;          // over B*N
  int b = rowBase >> 12;
  int nBase = rowBase & 4095;

  // stage u[b] (16 heads x 1024 k, f16) into LDS, padded stride
  {
    int h = tid >> 4;
    int ks = (tid & 15) * 64;
    const _Float16* src = u_t + ((size_t)(b * 16 + h)) * Cn + ks;
    _Float16* dst = &Us[h * USTRIDE + ks];
#pragma unroll
    for (int qq = 0; qq < 8; ++qq)
      *(v8h*)(dst + qq * 8) = *(const v8h*)(src + qq * 8);
  }
  __syncthreads();

  int wave = tid >> 5, lane = tid & 31;
  int lhalf = lane & 15;
  int koff = (lane < 16) ? 0 : 8;
  int rowW = rowBase + wave * 16;

  v8f acc = {};
  const float* xr0 = x + (size_t)(rowW + lhalf) * Cn + koff;
  for (int k0 = 0; k0 < Cn; k0 += 32) {
    const float* xr = xr0 + k0;
    v4f f0 = *(const v4f*)(xr);
    v4f f1 = *(const v4f*)(xr + 4);
    v4f f2 = *(const v4f*)(xr + 16);
    v4f f3 = *(const v4f*)(xr + 20);
    v16h a;
    v4h h0 = cvt4(f0), h1 = cvt4(f1), h2 = cvt4(f2), h3 = cvt4(f3);
#pragma unroll
    for (int i = 0; i < 4; ++i) {
      a[i] = h0[i]; a[i + 4] = h1[i]; a[i + 8] = h2[i]; a[i + 12] = h3[i];
    }
    v16h bb = mk16(&Us[lhalf * USTRIDE + k0 + koff]);
    acc = wmma_f16(a, bb, acc);
  }

  float betaV = beta[b * 16 + lhalf];
  int nLoc = nBase + wave * 16 + ((lane >> 4) << 3);
  float* srow = s + ((size_t)(b * 16 + lhalf)) * Nn;
#pragma unroll
  for (int vi = 0; vi < 8; ++vi) srow[nLoc + vi] = acc[vi] + betaV;
}

// -------------------------------------------- softmax over 4096 per (b,h) row
__global__ __launch_bounds__(256) void softmax_kernel(
    const float* __restrict__ s, float* __restrict__ attn) {
  __shared__ float red[256];
  int row = blockIdx.x;          // 256 rows
  int tid = threadIdx.x;
  const float* sr = s + (size_t)row * Nn;
  float* ar = attn + (size_t)row * Nn;

  float v[16];
  float m = -3.4e38f;
#pragma unroll
  for (int i = 0; i < 16; ++i) {
    v[i] = sr[tid + i * 256];
    m = fmaxf(m, v[i]);
  }
  red[tid] = m;
  __syncthreads();
  for (int st = 128; st > 0; st >>= 1) {
    if (tid < st) red[tid] = fmaxf(red[tid], red[tid + st]);
    __syncthreads();
  }
  m = red[0];
  __syncthreads();

  float sum = 0.f;
#pragma unroll
  for (int i = 0; i < 16; ++i) {
    v[i] = __expf(v[i] - m);
    sum += v[i];
  }
  red[tid] = sum;
  __syncthreads();
  for (int st = 128; st > 0; st >>= 1) {
    if (tid < st) red[tid] += red[tid + st];
    __syncthreads();
  }
  float inv = 1.0f / red[0];
#pragma unroll
  for (int i = 0; i < 16; ++i) ar[tid + i * 256] = v[i] * inv;
}

// -------------------------------------------- out = attn * (x@Wv + bv), 128x128 tiles
#define BM 128
#define BN 128
#define BK 32
#define LDA 40
__global__ __launch_bounds__(256) void vproj_kernel(
    const float* __restrict__ x, const _Float16* __restrict__ Wvt,
    const float* __restrict__ bv, const float* __restrict__ attn,
    float* __restrict__ out) {
  __shared__ __attribute__((aligned(16))) _Float16 As[BM * LDA];
  __shared__ __attribute__((aligned(16))) _Float16 Bs[2][BN * BK];

  int tid = threadIdx.x;
  int rowBase = blockIdx.x * BM;           // over B*N = 65536
  int colBase = blockIdx.y * BN;           // over C = 1024
  int wave = tid >> 5, lane = tid & 31;
  int waveM = (wave >> 1) * 32;
  int waveN = (wave & 1) * 64;
  int lhalf = lane & 15;
  int koff = (lane < 16) ? 0 : 8;

  const _Float16* wvtBase = Wvt + (size_t)colBase * Cn;

#if USE_TDM
  unsigned bs0 = (unsigned)(unsigned long long)(size_t)&Bs[0][0];
  if (wave == 0) {
    tdm_load_tile(bs0, (unsigned long long)(size_t)wvtBase, (unsigned)BK,
                  (unsigned)BN, (unsigned)Cn);
  }
#endif

  v8f acc[2][4] = {};
  int cur = 0;

  for (int k0 = 0; k0 < Cn; k0 += BK) {
#if USE_TDM
    if (wave == 0 && k0 + BK < Cn) {
      tdm_load_tile(bs0 + (unsigned)((cur ^ 1) * BN * BK * 2),
                    (unsigned long long)(size_t)(wvtBase + k0 + BK),
                    (unsigned)BK, (unsigned)BN, (unsigned)Cn);
    }
#endif
    // stage A: x[rowBase..+128][k0..+32] f32 -> f16 LDS (row major, padded)
    {
      int r = tid >> 3;
      int c4 = (tid & 7) * 4;
#pragma unroll
      for (int p = 0; p < 4; ++p, r += 32) {
        v4f f = *(const v4f*)(x + (size_t)(rowBase + r) * Cn + k0 + c4);
        *(v4h*)&As[r * LDA + c4] = cvt4(f);
      }
    }
#if USE_TDM
    if (wave == 0) {
      if (k0 + BK < Cn) __builtin_amdgcn_s_wait_tensorcnt(1);
      else              __builtin_amdgcn_s_wait_tensorcnt(0);
    }
#else
    // stage B: copy pre-transposed f16 slab Wvt[colBase..+128][k0..+32]
    {
      int col = tid >> 1;
      int hs = (tid & 1) * 16;
      const _Float16* src = wvtBase + (size_t)col * Cn + k0 + hs;
      _Float16* dst = &Bs[0][col * BK + hs];
      *(v8h*)dst = *(const v8h*)src;
      *(v8h*)(dst + 8) = *(const v8h*)(src + 8);
    }
#endif
    __syncthreads();

    v16h a[2], b[4];
#pragma unroll
    for (int i = 0; i < 2; ++i)
      a[i] = mk16(&As[(waveM + i * 16 + lhalf) * LDA + koff]);
#pragma unroll
    for (int j = 0; j < 4; ++j)
      b[j] = mk16(&Bs[cur][(waveN + j * 16 + lhalf) * BK + koff]);
#pragma unroll
    for (int i = 0; i < 2; ++i)
#pragma unroll
      for (int j = 0; j < 4; ++j)
        acc[i][j] = wmma_f16(a[i], b[j], acc[i][j]);
    __syncthreads();
#if USE_TDM
    cur ^= 1;
#endif
  }

  // epilogue: out[row, col] = (acc + bv[col]) * attn[b, col/64, n]
  int bb = rowBase >> 12;                  // /4096
  int nBase = rowBase & 4095;
  int h = (colBase + waveN) >> 6;          // wave's 64 cols are one head
  const float* attnRow = attn + ((size_t)(bb * 16 + h)) * Nn;

  float bvv[4];
#pragma unroll
  for (int j = 0; j < 4; ++j) bvv[j] = bv[colBase + waveN + j * 16 + lhalf];

#pragma unroll
  for (int i = 0; i < 2; ++i) {
    int rLoc = waveM + i * 16 + ((lane >> 4) << 3);
    float at[8];
#pragma unroll
    for (int vi = 0; vi < 8; ++vi) at[vi] = attnRow[nBase + rLoc + vi];
#pragma unroll
    for (int j = 0; j < 4; ++j) {
      int col = colBase + waveN + j * 16 + lhalf;
#pragma unroll
      for (int vi = 0; vi < 8; ++vi) {
        float val = (acc[i][j][vi] + bvv[j]) * at[vi];
        out[(size_t)(rowBase + rLoc + vi) * Cn + col] = val;
      }
    }
  }
}

extern "C" void kernel_launch(void* const* d_in, const int* in_sizes, int n_in,
                              void* d_out, int out_size, void* d_ws, size_t ws_size,
                              hipStream_t stream) {
  const float* x    = (const float*)d_in[0];
  const float* fore = (const float*)d_in[1];
  const float* Wq   = (const float*)d_in[2];
  const float* bq   = (const float*)d_in[3];
  const float* Wk   = (const float*)d_in[4];
  const float* bk   = (const float*)d_in[5];
  const float* Wv   = (const float*)d_in[6];
  const float* bv   = (const float*)d_in[7];

  float* out     = (float*)d_out;
  float* attnOut = out + (size_t)Bn * Nn * Cn;     // second output region

  char* ws = (char*)d_ws;
  float*     q    = (float*)(ws);                  // 16*1024 f32      = 64 KB
  float*     beta = (float*)(ws + 65536);          // 256 f32
  _Float16*  u_t  = (_Float16*)(ws + 66560);       // 16*16*1024 f16   = 512 KB
  float*     s    = (float*)(ws + 590848);         // 16*16*4096 f32   = 4 MB
  _Float16*  Wvt  = (_Float16*)(ws + 4785152);     // 1024*1024 f16    = 2 MB

  qproj_kernel   <<<64,   256, 0, stream>>>(fore, Wq, bq, q);
  uproj_kernel   <<<1024, 256, 0, stream>>>(Wk, q, u_t);
  betaproj_kernel<<<1,    256, 0, stream>>>(bk, q, beta);
  wvt_kernel     <<<dim3(32, 32), 256, 0, stream>>>(Wv, Wvt);
  score_kernel   <<<512,  256, 0, stream>>>(x, u_t, beta, s);
  softmax_kernel <<<256,  256, 0, stream>>>(s, attnOut);
  vproj_kernel   <<<dim3(512, 8), 256, 0, stream>>>(x, Wvt, bv, attnOut, out);
}
